// SelectiveSSM_24739011625084
// MI455X (gfx1250) — matrix operand order repaired and checked by reading, not compiled, not run on hardware
//
#include <hip/hip_runtime.h>

// ---------------------------------------------------------------------------
// SelectiveSSM (Mamba-style block) for gfx1250.
// B=2, T=2048, D=768, N=16.  GEMMs in bf16 WMMA (f32 accum), scan chunked.
// ---------------------------------------------------------------------------

typedef __attribute__((ext_vector_type(16))) __bf16 v16bf;
typedef __attribute__((ext_vector_type(8)))  __bf16 v8bf;
typedef __attribute__((ext_vector_type(8)))  float  v8f;

static constexpr int kB  = 2;
static constexpr int kT  = 2048;
static constexpr int kD  = 768;
static constexpr int kN  = 16;
static constexpr int kM  = kB * kT;          // 4096 rows
static constexpr int kChunks = 16;           // scan chunks
static constexpr int kL  = kT / kChunks;     // 128 steps per chunk

// ---------------------------------------------------------------------------
// Weight conversion fp32 -> bf16 (and pack B_w/C_w into one 32xD matrix)
// ---------------------------------------------------------------------------
__global__ void convert_weights_kernel(
    const float* __restrict__ ipw, const float* __restrict__ dtw,
    const float* __restrict__ outw, const float* __restrict__ Bw,
    const float* __restrict__ Cw,
    __bf16* __restrict__ ipb, __bf16* __restrict__ dtb,
    __bf16* __restrict__ outb, __bf16* __restrict__ bcb)
{
    const int stride = gridDim.x * blockDim.x;
    const int tid0 = blockIdx.x * blockDim.x + threadIdx.x;
    for (int i = tid0; i < 2 * kD * kD; i += stride) ipb[i]  = (__bf16)ipw[i];
    for (int i = tid0; i < kD * kD;     i += stride) dtb[i]  = (__bf16)dtw[i];
    for (int i = tid0; i < kD * kD;     i += stride) outb[i] = (__bf16)outw[i];
    for (int i = tid0; i < 2 * kN * kD; i += stride) {
        int row = i / kD, col = i % kD;
        float v = (row < kN) ? Bw[row * kD + col] : Cw[(row - kN) * kD + col];
        bcb[i] = (__bf16)v;
    }
}

// ---------------------------------------------------------------------------
// LayerNorm over D, output bf16 (GEMM1 A operand).  One block per (b,t) row.
// ---------------------------------------------------------------------------
__global__ __launch_bounds__(256) void layernorm_kernel(
    const float* __restrict__ x, const float* __restrict__ w,
    const float* __restrict__ bvec, __bf16* __restrict__ out)
{
    __shared__ float red[256];
    const int row = blockIdx.x;
    const int tid = threadIdx.x;
    const float* xr = x + (size_t)row * kD;
    float v0 = xr[tid], v1 = xr[tid + 256], v2 = xr[tid + 512];
    red[tid] = v0 + v1 + v2;
    __syncthreads();
    for (int o = 128; o > 0; o >>= 1) {
        if (tid < o) red[tid] += red[tid + o];
        __syncthreads();
    }
    const float mu = red[0] * (1.0f / kD);
    __syncthreads();
    float d0 = v0 - mu, d1 = v1 - mu, d2 = v2 - mu;
    red[tid] = d0 * d0 + d1 * d1 + d2 * d2;
    __syncthreads();
    for (int o = 128; o > 0; o >>= 1) {
        if (tid < o) red[tid] += red[tid + o];
        __syncthreads();
    }
    const float rstd = rsqrtf(red[0] * (1.0f / kD) + 1e-5f);
    __bf16* orow = out + (size_t)row * kD;
    orow[tid]       = (__bf16)(d0 * rstd * w[tid]       + bvec[tid]);
    orow[tid + 256] = (__bf16)(d1 * rstd * w[tid + 256] + bvec[tid + 256]);
    orow[tid + 512] = (__bf16)(d2 * rstd * w[tid + 512] + bvec[tid + 512]);
}

// ---------------------------------------------------------------------------
// bf16 WMMA GEMM:  out[M x N] = A[M x K] * W[N x K]^T  (+ epilogue)
// Each wave computes a (16*MT) x (16*TN) tile; K stepped by 32, manually
// unrolled 2x with ping-pong fragment buffers (no register copies, loads of
// step k+1 overlap WMMAs of step k).  All 8 waves of a block share one
// N-strip (same B addresses -> WGP$ hits), differing in M.
// Requires K % 64 == 0.
// MODE: 0 = plain store, 1 = +bias then softplus, 2 = +residual.
//
// A fragment layout (16x32 bf16): lane half=lane>>4 selects K-halves
//   elems 0..7  -> K = k0 + 8*half + e ;  elems 8..15 -> K = k0+16+8*half+(e-8)
// B operand = rows of W (W is N x K row-major == (K x N)^T operand):
//   lane n=lane&15 col, same K mapping as A fragment
// D element r -> (M = 8*half + r, N = lane&15)
// ---------------------------------------------------------------------------
template <int MT, int TN, int MODE>
__global__ __launch_bounds__(256) void gemm_bf16_wmma(
    const __bf16* __restrict__ A, const __bf16* __restrict__ W,
    const float* __restrict__ bias, const float* __restrict__ resid,
    float* __restrict__ out, int M, int N, int K, int ldc)
{
    const int lane = threadIdx.x & 31;
    const int wave = threadIdx.x >> 5;
    const int tilesN = N / (16 * TN);
    const int nTile  = blockIdx.x % tilesN;
    const int mGroup = blockIdx.x / tilesN;           // 8*MT m-tiles per block
    const int mTile0 = (mGroup * 8 + wave) * MT;

    const int half  = lane >> 4;
    const int l16   = lane & 15;
    const int ncol0 = nTile * (16 * TN);

    v8f acc[MT][TN];
#pragma unroll
    for (int mt = 0; mt < MT; ++mt)
#pragma unroll
        for (int tn = 0; tn < TN; ++tn)
#pragma unroll
            for (int i = 0; i < 8; ++i) acc[mt][tn][i] = 0.0f;

    auto loadAB = [&](v16bf (&a)[MT], v16bf (&b)[TN], int k0) {
#pragma unroll
        for (int mt = 0; mt < MT; ++mt) {
            const __bf16* ap =
                A + (size_t)((mTile0 + mt) * 16 + l16) * K + k0 + half * 8;
            v8bf lo = *(const v8bf*)(ap);
            v8bf hi = *(const v8bf*)(ap + 16);
#pragma unroll
            for (int i = 0; i < 8; ++i) { a[mt][i] = lo[i]; a[mt][i + 8] = hi[i]; }
        }
#pragma unroll
        for (int tn = 0; tn < TN; ++tn) {
            const __bf16* bp =
                W + (size_t)(ncol0 + tn * 16 + l16) * K + k0 + half * 8;
            v8bf b0 = *(const v8bf*)(bp);
            v8bf b1 = *(const v8bf*)(bp + 16);
#pragma unroll
            for (int i = 0; i < 8; ++i) { b[tn][i] = b0[i]; b[tn][i + 8] = b1[i]; }
        }
    };
    auto mma = [&](v16bf (&a)[MT], v16bf (&b)[TN]) {
#pragma unroll
        for (int mt = 0; mt < MT; ++mt)
#pragma unroll
            for (int tn = 0; tn < TN; ++tn)
                acc[mt][tn] = __builtin_amdgcn_wmma_f32_16x16x32_bf16(
                    false, a[mt], false, b[tn], (short)0, acc[mt][tn],
                    false, false);
    };

    v16bf a0[MT], b0f[TN], a1[MT], b1f[TN];
    loadAB(a0, b0f, 0);
    int k0 = 0;
    for (; k0 + 64 < K; k0 += 64) {        // steady state: no branches/copies
        loadAB(a1, b1f, k0 + 32);
        mma(a0, b0f);
        loadAB(a0, b0f, k0 + 64);
        mma(a1, b1f);
    }
    loadAB(a1, b1f, k0 + 32);              // tail: last two k-steps
    mma(a0, b0f);
    mma(a1, b1f);

#pragma unroll
    for (int mt = 0; mt < MT; ++mt) {
        const int mbase = (mTile0 + mt) * 16 + half * 8;
#pragma unroll
        for (int tn = 0; tn < TN; ++tn) {
            const int nn = ncol0 + tn * 16 + l16;
            const float bval = (MODE == 1) ? bias[nn] : 0.0f;
#pragma unroll
            for (int r = 0; r < 8; ++r) {
                const int mm = mbase + r;
                float v = acc[mt][tn][r];
                if (MODE == 1) {                 // bias + softplus
                    v += bval;
                    v = (v > 20.0f) ? v : log1pf(expf(v));
                }
                if (MODE == 2) v += resid[(size_t)mm * ldc + nn];
                out[(size_t)mm * ldc + nn] = v;
            }
        }
    }
}

// ---------------------------------------------------------------------------
// Depthwise causal conv1d (K=4, left pad 3) + SiLU.  Writes f32 + bf16 copies.
// x_ssm lives in the first D columns of the xz buffer (row stride 2D).
// ---------------------------------------------------------------------------
__global__ __launch_bounds__(256) void conv_silu_kernel(
    const float* __restrict__ xz, const float* __restrict__ cw,
    const float* __restrict__ cb, float* __restrict__ xact_f,
    __bf16* __restrict__ xact_b)
{
    const int idx = blockIdx.x * blockDim.x + threadIdx.x;  // < M*D
    const int d  = idx % kD;
    const int bt = idx / kD;
    const int t  = bt % kT;
    float acc = cb[d];
#pragma unroll
    for (int j = 0; j < 4; ++j) {
        const int ts = t - 3 + j;
        if (ts >= 0)
            acc += cw[d * 4 + j] * xz[(size_t)(bt - 3 + j) * (2 * kD) + d];
    }
    const float s = acc / (1.0f + expf(-acc));   // SiLU
    xact_f[idx] = s;
    xact_b[idx] = (__bf16)s;
}

// ---------------------------------------------------------------------------
// Selective scan, 3-phase chunked.
// Recurrence:  h_t = exp(clamp(dt_t*A, +-20)) * h_{t-1} + (dt_t*x_t) * B_t
//              y_t = <h_t, C_t>
// Phase A: per (b,d,chunk) compute chunk decay product E and local state U.
// Phase B: per (b,d) sequential combine over chunks -> chunk start states H0.
// Phase C: replay chunk from H0, emit y2 = (y + x*D)*silu(z)  (bf16).
// ---------------------------------------------------------------------------
__global__ __launch_bounds__(256) void scan_phaseA_kernel(
    const float* __restrict__ dt, const float* __restrict__ xact,
    const float* __restrict__ bc, const float* __restrict__ logA,
    float* __restrict__ E, float* __restrict__ U)
{
    __shared__ float sB[kL][kN];
    const int tid   = threadIdx.x;
    const int cell  = blockIdx.x / (kD / 256);
    const int dblk  = blockIdx.x % (kD / 256);
    const int b = cell / kChunks, c = cell % kChunks;
    const int d = dblk * 256 + tid;
    const int t0 = c * kL;
    for (int i = tid; i < kL * kN; i += 256) {
        const int tt = i >> 4, j = i & 15;
        sB[tt][j] = bc[(size_t)(b * kT + t0 + tt) * (2 * kN) + j];
    }
    __syncthreads();
    const float Ad = -expf(logA[d]);
    float h[kN];
#pragma unroll
    for (int n = 0; n < kN; ++n) h[n] = 0.0f;
    float sla = 0.0f;
    for (int tt = 0; tt < kL; ++tt) {
        const size_t gi = (size_t)(b * kT + t0 + tt) * kD + d;
        const float dtv = dt[gi], xv = xact[gi];
        const float la = fminf(fmaxf(dtv * Ad, -20.0f), 20.0f);
        const float e = expf(la);
        sla += la;
        const float dtx = dtv * xv;
#pragma unroll
        for (int n = 0; n < kN; ++n) h[n] = e * h[n] + dtx * sB[tt][n];
    }
    const size_t ub = ((size_t)(b * kChunks + c) * kD + d) * kN;
#pragma unroll
    for (int n = 0; n < kN; ++n) U[ub + n] = h[n];
    E[(size_t)(b * kChunks + c) * kD + d] = expf(sla);
}

__global__ __launch_bounds__(256) void scan_phaseB_kernel(
    const float* __restrict__ E, const float* __restrict__ U,
    float* __restrict__ H0)
{
    const int idx = blockIdx.x * 256 + threadIdx.x;   // < B*D
    const int b = idx / kD, d = idx % kD;
    float h[kN];
#pragma unroll
    for (int n = 0; n < kN; ++n) h[n] = 0.0f;
    for (int c = 0; c < kChunks; ++c) {
        const size_t base = ((size_t)(b * kChunks + c) * kD + d) * kN;
#pragma unroll
        for (int n = 0; n < kN; ++n) H0[base + n] = h[n];
        const float e = E[(size_t)(b * kChunks + c) * kD + d];
#pragma unroll
        for (int n = 0; n < kN; ++n) h[n] = e * h[n] + U[base + n];
    }
}

__global__ __launch_bounds__(256) void scan_phaseC_kernel(
    const float* __restrict__ dt, const float* __restrict__ xact,
    const float* __restrict__ bc, const float* __restrict__ xz,
    const float* __restrict__ H0, const float* __restrict__ logA,
    const float* __restrict__ Dp, __bf16* __restrict__ y2)
{
    __shared__ float sB[kL][kN];
    __shared__ float sC[kL][kN];
    const int tid  = threadIdx.x;
    const int cell = blockIdx.x / (kD / 256);
    const int dblk = blockIdx.x % (kD / 256);
    const int b = cell / kChunks, c = cell % kChunks;
    const int d = dblk * 256 + tid;
    const int t0 = c * kL;
    for (int i = tid; i < kL * 2 * kN; i += 256) {
        const int tt = i >> 5, j = i & 31;
        const float v = bc[(size_t)(b * kT + t0 + tt) * (2 * kN) + j];
        if (j < kN) sB[tt][j] = v; else sC[tt][j - kN] = v;
    }
    __syncthreads();
    const float Ad  = -expf(logA[d]);
    const float Dpd = Dp[d];
    float h[kN];
    const size_t hb = ((size_t)(b * kChunks + c) * kD + d) * kN;
#pragma unroll
    for (int n = 0; n < kN; ++n) h[n] = H0[hb + n];
    for (int tt = 0; tt < kL; ++tt) {
        const size_t gi = (size_t)(b * kT + t0 + tt) * kD + d;
        const float dtv = dt[gi], xv = xact[gi];
        const float la = fminf(fmaxf(dtv * Ad, -20.0f), 20.0f);
        const float e = expf(la);
        const float dtx = dtv * xv;
        float y = 0.0f;
#pragma unroll
        for (int n = 0; n < kN; ++n) {
            h[n] = e * h[n] + dtx * sB[tt][n];
            y += h[n] * sC[tt][n];
        }
        const float zv = xz[(size_t)(b * kT + t0 + tt) * (2 * kD) + kD + d];
        const float gate = zv / (1.0f + expf(-zv));      // silu(z)
        y2[gi] = (__bf16)((y + xv * Dpd) * gate);
    }
}

// ---------------------------------------------------------------------------
// Host-side orchestration
// ---------------------------------------------------------------------------
extern "C" void kernel_launch(void* const* d_in, const int* in_sizes, int n_in,
                              void* d_out, int out_size, void* d_ws, size_t ws_size,
                              hipStream_t stream)
{
    const float* x      = (const float*)d_in[0];
    const float* norm_w = (const float*)d_in[1];
    const float* norm_b = (const float*)d_in[2];
    const float* ipw    = (const float*)d_in[3];
    const float* convw  = (const float*)d_in[4];
    const float* convb  = (const float*)d_in[5];
    const float* dtw    = (const float*)d_in[6];
    const float* dtb    = (const float*)d_in[7];
    const float* Bw     = (const float*)d_in[8];
    const float* Cw     = (const float*)d_in[9];
    const float* outw   = (const float*)d_in[10];
    const float* logA   = (const float*)d_in[11];
    const float* Dp     = (const float*)d_in[12];
    float* out = (float*)d_out;

    // workspace carve-up (256B aligned)
    char* ws = (char*)d_ws;
    size_t off = 0;
    auto take = [&](size_t bytes) -> char* {
        char* p = ws + off;
        off = (off + bytes + 255) & ~(size_t)255;
        return p;
    };
    __bf16* xn_bf   = (__bf16*)take((size_t)kM * kD * 2);            // LN output
    float*  xz_f    = (float*) take((size_t)kM * 2 * kD * 4);        // in_proj out
    float*  xact_f  = (float*) take((size_t)kM * kD * 4);            // conv+silu
    __bf16* xact_bf = (__bf16*)take((size_t)kM * kD * 2);
    float*  dt_f    = (float*) take((size_t)kM * kD * 4);            // softplus dt
    float*  bc_f    = (float*) take((size_t)kM * 2 * kN * 4);        // [B|C] rows
    __bf16* y2_bf   = (__bf16*)take((size_t)kM * kD * 2);            // gated y
    __bf16* wip_bf  = (__bf16*)take((size_t)2 * kD * kD * 2);
    __bf16* wdt_bf  = (__bf16*)take((size_t)kD * kD * 2);
    __bf16* wout_bf = (__bf16*)take((size_t)kD * kD * 2);
    __bf16* wbc_bf  = (__bf16*)take((size_t)2 * kN * kD * 2);
    float*  E_f     = (float*) take((size_t)kB * kChunks * kD * 4);
    float*  U_f     = (float*) take((size_t)kB * kChunks * kD * kN * 4);
    float*  H0_f    = (float*) take((size_t)kB * kChunks * kD * kN * 4);

    // 1) weights -> bf16
    convert_weights_kernel<<<1024, 256, 0, stream>>>(
        ipw, dtw, outw, Bw, Cw, wip_bf, wdt_bf, wout_bf, wbc_bf);

    // 2) LayerNorm (one block per row)
    layernorm_kernel<<<kM, 256, 0, stream>>>(x, norm_w, norm_b, xn_bf);

    // 3) GEMM1: xz = xn @ in_proj_w.T   (4096 x 1536 x 768)
    {
        const int tilesN = (2 * kD) / 64;             // 24
        const int blocks = tilesN * (kM / 16 / 16);   // 24 * 16 = 384
        gemm_bf16_wmma<2, 4, 0><<<blocks, 256, 0, stream>>>(
            xn_bf, wip_bf, nullptr, nullptr, xz_f, kM, 2 * kD, kD, 2 * kD);
    }

    // 4) depthwise conv + SiLU
    conv_silu_kernel<<<(kM * kD) / 256, 256, 0, stream>>>(
        xz_f, convw, convb, xact_f, xact_bf);

    // 5) GEMM2: dt = softplus(xact @ dt_w.T + dt_b)   (4096 x 768 x 768)
    {
        const int tilesN = kD / 64;                   // 12
        const int blocks = tilesN * (kM / 16 / 16);   // 12 * 16 = 192
        gemm_bf16_wmma<2, 4, 1><<<blocks, 256, 0, stream>>>(
            xact_bf, wdt_bf, dtb, nullptr, dt_f, kM, kD, kD, kD);
    }

    // 6) GEMM3: [B_mat | C_mat] = xact @ [B_w;C_w].T   (4096 x 32 x 768)
    {
        const int blocks = kM / 16 / 16;              // tilesN == 1 -> 16
        gemm_bf16_wmma<2, 2, 0><<<blocks, 256, 0, stream>>>(
            xact_bf, wbc_bf, nullptr, nullptr, bc_f, kM, 2 * kN, kD, 2 * kN);
    }

    // 7) chunked selective scan
    scan_phaseA_kernel<<<kB * kChunks * (kD / 256), 256, 0, stream>>>(
        dt_f, xact_f, bc_f, logA, E_f, U_f);
    scan_phaseB_kernel<<<(kB * kD) / 256, 256, 0, stream>>>(E_f, U_f, H0_f);
    scan_phaseC_kernel<<<kB * kChunks * (kD / 256), 256, 0, stream>>>(
        dt_f, xact_f, bc_f, xz_f, H0_f, logA, Dp, y2_bf);

    // 8) GEMM4: out = x + y2 @ out_w.T   (4096 x 768 x 768, residual add)
    {
        const int tilesN = kD / 64;                   // 12
        const int blocks = tilesN * (kM / 16 / 16);   // 192
        gemm_bf16_wmma<2, 4, 2><<<blocks, 256, 0, stream>>>(
            y2_bf, wout_bf, nullptr, x, out, kM, kD, kD, kD);
    }
}
